// ModelSingleStep_42898133353466
// MI455X (gfx1250) — matrix operand order repaired and checked by reading, not compiled, not run on hardware
//
#include <hip/hip_runtime.h>
#include <cstddef>

// ---------------------------------------------------------------------------
// CDNA5 / gfx1250 implementation.
//  - All dense layers as bf16 WMMA GEMMs (v_wmma_f32_16x16x32_bf16),
//    register-blocked 64(M)x16(N) per wave: 4 WMMAs per B-fragment load.
//  - LSTM recurrence as a single-workgroup scan with W_hh resident in the
//    320KB WGP LDS (bf16: exactly 320,000 bytes).
// ---------------------------------------------------------------------------

typedef __bf16 bf16_t;
typedef __attribute__((ext_vector_type(16))) __bf16 v16bf;
typedef __attribute__((ext_vector_type(8)))  __bf16 v8bf;
typedef __attribute__((ext_vector_type(8)))  float  v8f;

#define DEV_INLINE __device__ __forceinline__

DEV_INLINE float dsig(float x)   { return 1.0f / (1.0f + __expf(-x)); }
DEV_INLINE float dlrelu(float x) { return x >= 0.0f ? x : 0.01f * x; }

union BF16V { v16bf v; v8bf h[2]; };

// ------------------------- conversion / prep kernels -----------------------

// X[K,N] (f32, row-major) -> Xt[N,Kp] (bf16, frame-major, zero K-padding)
__global__ void k_transpose_x_bf16(const float* __restrict__ X,
                                   bf16_t* __restrict__ Xt,
                                   int K, int Kp, int N)
{
    size_t i = (size_t)blockIdx.x * blockDim.x + threadIdx.x;
    size_t total = (size_t)N * (size_t)Kp;
    if (i >= total) return;
    int n = (int)(i / (size_t)Kp);
    int k = (int)(i % (size_t)Kp);
    Xt[i] = (k < K) ? (bf16_t)X[(size_t)k * (size_t)N + n] : (bf16_t)0.0f;
}

// W[M,K] f32 -> Wb[Mp,Kp] bf16 with zero padding rows/cols
__global__ void k_pad_weight_bf16(const float* __restrict__ W,
                                  bf16_t* __restrict__ Wb,
                                  int M, int K, int Mp, int Kp)
{
    size_t i = (size_t)blockIdx.x * blockDim.x + threadIdx.x;
    size_t total = (size_t)Mp * (size_t)Kp;
    if (i >= total) return;
    int m = (int)(i / (size_t)Kp);
    int k = (int)(i % (size_t)Kp);
    Wb[i] = (m < M && k < K) ? (bf16_t)W[(size_t)m * (size_t)K + k] : (bf16_t)0.0f;
}

__global__ void k_bias_sum(const float* __restrict__ a, const float* __restrict__ b,
                           float* __restrict__ o, int n)
{
    int i = blockIdx.x * blockDim.x + threadIdx.x;
    if (i < n) o[i] = a[i] + b[i];
}

// ------------------------------ WMMA mainloop ------------------------------
// ISA 16-bit fragment layouts (wave32):
//   A 16x32: lanes 0-15 rows M, K chunks {0..7,16..23}; lanes 16-31 {8..15,24..31}
//   B 32x16: lanes 0-15 cols N, K chunks {0..7,16..23}; lanes 16-31 {8..15,24..31}
//   C/D f32 16x16: lane (hi=lane>>4, col=lane&15) holds col n0+col,
//                  rows m0 + hi*8 + v for VGPR v in 0..7.
//
// Each wave computes a 64(M) x 16(N) macro-tile: one B fragment feeds 4 WMMAs.
// Partial M handled by clamping the A row base to the last valid 16-row tile
// (loads in-bounds, EXEC all-1s) and skipping the store for invalid tiles.

struct AccTile { v8f a[4]; };

DEV_INLINE AccTile wmma_kloop4(const bf16_t* __restrict__ W, const bf16_t* __restrict__ B,
                               int Kp, int ldb, int Mp, int m0, int n0, int lane)
{
    AccTile acc;
#pragma unroll
    for (int t = 0; t < 4; ++t) acc.a[t] = (v8f){0.f,0.f,0.f,0.f,0.f,0.f,0.f,0.f};

    const int half = lane >> 4;      // 0: lanes 0-15, 1: lanes 16-31
    const int low  = lane & 15;
    const bf16_t* wr[4];
#pragma unroll
    for (int t = 0; t < 4; ++t) {
        int m = m0 + t * 16;
        int mc = (m < Mp) ? m : (Mp - 16);           // clamp: stay in-bounds
        wr[t] = W + (size_t)(mc + low) * (size_t)Kp + (half ? 8 : 0);
    }
    const bf16_t* br = B + (size_t)(n0 + low) * (size_t)ldb + (half ? 8 : 0);

    for (int k = 0; k < Kp; k += 32) {
        BF16V b;
        b.h[0] = *(const v8bf*)(br + k);
        b.h[1] = *(const v8bf*)(br + k + 16);
#pragma unroll
        for (int t = 0; t < 4; ++t) {
            BF16V a;
            a.h[0] = *(const v8bf*)(wr[t] + k);
            a.h[1] = *(const v8bf*)(wr[t] + k + 16);
            acc.a[t] = __builtin_amdgcn_wmma_f32_16x16x32_bf16(
                false, a.v, false, b.v, (short)0, acc.a[t], false, false);
        }
    }
    return acc;
}

template <bool F32OUT, bool ACT_LRELU>
__global__ void __launch_bounds__(256, 2)
k_gemm_wmma(const bf16_t* __restrict__ W, const bf16_t* __restrict__ B,
            const float* __restrict__ bias, int Mreal, int Mp, int Kp,
            int ldb, int ldc, int colOff,
            bf16_t* __restrict__ outB, float* __restrict__ outF)
{
    const int lane = threadIdx.x & 31;
    const int wave = threadIdx.x >> 5;
    const int n0 = (blockIdx.x * 8 + wave) * 16;   // 8 waves tile along N
    const int m0 = blockIdx.y * 64;                // 4 M-tiles per wave

    AccTile acc = wmma_kloop4(W, B, Kp, ldb, Mp, m0, n0, lane);

    const int half  = lane >> 4;
    const int frame = n0 + (lane & 15);
#pragma unroll
    for (int t = 0; t < 4; ++t) {
        const int mt = m0 + t * 16;
        if (mt >= Mp) continue;                    // wave-uniform
        const int mb = mt + half * 8;
#pragma unroll
        for (int v = 0; v < 8; ++v) {
            const int m = mb + v;
            float x = acc.a[t][v] + ((m < Mreal) ? bias[m] : 0.0f);
            if (ACT_LRELU) x = dlrelu(x);
            size_t idx = (size_t)frame * (size_t)ldc + (size_t)(colOff + m);
            if (F32OUT) outF[idx] = x;
            else        outB[idx] = (bf16_t)x;
        }
    }
}

// Final layer: mask = sigmoid(W4 @ d + b4); out[m,n] = mask * X[m,n] (f32).
__global__ void __launch_bounds__(256, 2)
k_gemm_mask_out(const bf16_t* __restrict__ W, const bf16_t* __restrict__ B,
                const float* __restrict__ bias, const float* __restrict__ X,
                float* __restrict__ out, int Mreal, int Mp, int Kp, int ldb, int N)
{
    const int lane = threadIdx.x & 31;
    const int wave = threadIdx.x >> 5;
    const int n0 = (blockIdx.x * 8 + wave) * 16;
    const int m0 = blockIdx.y * 64;

    AccTile acc = wmma_kloop4(W, B, Kp, ldb, Mp, m0, n0, lane);

    const int half  = lane >> 4;
    const int frame = n0 + (lane & 15);
#pragma unroll
    for (int t = 0; t < 4; ++t) {
        const int mt = m0 + t * 16;
        if (mt >= Mp) continue;
        const int mb = mt + half * 8;
#pragma unroll
        for (int v = 0; v < 8; ++v) {
            const int m = mb + v;
            if (m < Mreal) {
                float s = dsig(acc.a[t][v] + bias[m]);
                size_t idx = (size_t)m * (size_t)N + frame;
                out[idx] = s * X[idx];
            }
        }
    }
}

// ------------------------------- LSTM scan ---------------------------------
// One workgroup, 800 threads (25 waves).  W_hh (800x200) lives in LDS as bf16
// (320,000 B) -- a CDNA5-only fit (320 KB LDS per WGP).  Per step: each thread
// computes one gate row; threads 0..199 apply the LSTM cell and publish h both
// to LDS and into HCAT[f, 0:200] (bf16, frame-major) for the fuse GEMM.
__global__ void __launch_bounds__(800)
k_lstm_scan(const float* __restrict__ Whh, const float* __restrict__ Gin,
            bf16_t* __restrict__ Hcat, int F)
{
    extern __shared__ char smem[];
    bf16_t* w     = (bf16_t*)smem;                       // 800*200 bf16 = 320000 B
    float*  h     = (float*)(smem + 320000);             // 200 f32
    float*  gates = (float*)(smem + 320000 + 1024);      // 800 f32

    const int tid = threadIdx.x;
    for (int i = tid; i < 800 * 200; i += 800) w[i] = (bf16_t)Whh[i];
    if (tid < 200) h[tid] = 0.0f;
    float cval = 0.0f;
    __syncthreads();

    const bf16_t* wr = w + tid * 200;
    for (int f = 0; f < F; ++f) {
        float g = Gin[(size_t)f * 800 + tid];
#pragma unroll 8
        for (int k = 0; k < 200; ++k) g += (float)wr[k] * h[k];
        gates[tid] = g;
        __syncthreads();
        if (tid < 200) {
            float iv = dsig(gates[tid]);
            float fv = dsig(gates[200 + tid]);
            float gv = tanhf(gates[400 + tid]);
            float ov = dsig(gates[600 + tid]);
            cval = fv * cval + iv * gv;
            float h1 = ov * tanhf(cval);
            h[tid] = h1;
            Hcat[(size_t)f * 608 + tid] = (bf16_t)h1;
        }
        __syncthreads();
    }
}

// -------------------------------- launcher ---------------------------------

extern "C" void kernel_launch(void* const* d_in, const int* in_sizes, int n_in,
                              void* d_out, int out_size, void* d_ws, size_t ws_size,
                              hipStream_t stream)
{
    (void)in_sizes; (void)n_in; (void)out_size; (void)ws_size;

    const float* X   = (const float*)d_in[0];   // [2049, 8192]
    const float* W1  = (const float*)d_in[1];   // [1000, 2049]
    const float* b1  = (const float*)d_in[2];
    const float* W2  = (const float*)d_in[3];   // [400, 1000]
    const float* b2  = (const float*)d_in[4];
    const float* W3  = (const float*)d_in[5];   // [1000, 400]
    const float* b3  = (const float*)d_in[6];
    const float* W4  = (const float*)d_in[7];   // [2049, 1000]
    const float* b4  = (const float*)d_in[8];
    const float* Wf1 = (const float*)d_in[9];   // [800, 600]
    const float* bf1 = (const float*)d_in[10];
    const float* Wf2 = (const float*)d_in[11];  // [400, 800]
    const float* bf2 = (const float*)d_in[12];
    const float* Wih = (const float*)d_in[13];  // [800, 400]
    const float* bih = (const float*)d_in[14];
    const float* Whh = (const float*)d_in[15];  // [800, 200]
    const float* bhh = (const float*)d_in[16];
    float* out = (float*)d_out;

    constexpr int F = 8192;

    // Workspace carve-out (~119 MB), 256B aligned slices.
    char* ws = (char*)d_ws;
    size_t off = 0;
    auto alloc = [&](size_t bytes) -> void* {
        void* p = ws + off;
        off += bytes;
        off = (off + 255) & ~(size_t)255;
        return p;
    };
    bf16_t* XT   = (bf16_t*)alloc((size_t)F * 2080 * 2);      // X^T bf16, Kpad
    bf16_t* WB1  = (bf16_t*)alloc((size_t)1008 * 2080 * 2);
    bf16_t* WB2  = (bf16_t*)alloc((size_t)416 * 1008 * 2);
    bf16_t* WBih = (bf16_t*)alloc((size_t)800 * 416 * 2);
    bf16_t* WBf1 = (bf16_t*)alloc((size_t)800 * 608 * 2);
    bf16_t* WBf2 = (bf16_t*)alloc((size_t)416 * 800 * 2);
    bf16_t* WB3  = (bf16_t*)alloc((size_t)1008 * 416 * 2);
    bf16_t* WB4  = (bf16_t*)alloc((size_t)2064 * 1008 * 2);
    bf16_t* E1   = (bf16_t*)alloc((size_t)F * 1008 * 2);      // layer-1 act; reused as D
    bf16_t* HCAT = (bf16_t*)alloc(((size_t)F * 608 + 64) * 2);// [F,608]: h|e|pad (+slack)
    float*  GIN  = (float*) alloc((size_t)F * 800 * 4);       // pre-recurrence gates
    bf16_t* TB   = (bf16_t*)alloc((size_t)F * 800 * 2);       // fuse layer 1 act
    bf16_t* T2   = (bf16_t*)alloc((size_t)F * 416 * 2);       // fuse layer 2 act
    float*  BSUM = (float*) alloc((size_t)800 * 4);           // b_ih + b_hh

    auto cdiv = [](size_t a, size_t b) { return (unsigned)((a + b - 1) / b); };

    // --- prep: transpose X to bf16 frame-major, convert/pad weights ---
    k_transpose_x_bf16<<<cdiv((size_t)F * 2080, 256), 256, 0, stream>>>(X, XT, 2049, 2080, F);
    k_pad_weight_bf16<<<cdiv((size_t)1008 * 2080, 256), 256, 0, stream>>>(W1,  WB1,  1000, 2049, 1008, 2080);
    k_pad_weight_bf16<<<cdiv((size_t)416  * 1008, 256), 256, 0, stream>>>(W2,  WB2,   400, 1000,  416, 1008);
    k_pad_weight_bf16<<<cdiv((size_t)800  * 416,  256), 256, 0, stream>>>(Wih, WBih,  800,  400,  800,  416);
    k_pad_weight_bf16<<<cdiv((size_t)800  * 608,  256), 256, 0, stream>>>(Wf1, WBf1,  800,  600,  800,  608);
    k_pad_weight_bf16<<<cdiv((size_t)416  * 800,  256), 256, 0, stream>>>(Wf2, WBf2,  400,  800,  416,  800);
    k_pad_weight_bf16<<<cdiv((size_t)1008 * 416,  256), 256, 0, stream>>>(W3,  WB3,  1000,  400, 1008,  416);
    k_pad_weight_bf16<<<cdiv((size_t)2064 * 1008, 256), 256, 0, stream>>>(W4,  WB4,  2049, 1000, 2064, 1008);
    k_bias_sum<<<4, 256, 0, stream>>>(bih, bhh, BSUM, 800);

    // Grid: block = 8 waves tiling 128 frames (N); blockIdx.y covers 64 rows (M).
    auto ggrid = [&](int Mp) { return dim3(F / 128, cdiv((size_t)Mp, 64)); };
    dim3 blk(256);

    // --- phase 1: frame-parallel encoder + gate pre-computation ---
    // E1[F,1008] = lrelu(W1 @ x + b1)
    k_gemm_wmma<false, true ><<<ggrid(1008), blk, 0, stream>>>(WB1, XT, b1, 1000, 1008, 2080, 2080, 1008, 0, E1, nullptr);
    // HCAT[F, 200:616] = lrelu(W2 @ e1 + b2)   (Mp=416 also zero-fills pad cols)
    k_gemm_wmma<false, true ><<<ggrid(416),  blk, 0, stream>>>(WB2, E1, b2, 400, 416, 1008, 1008, 608, 200, HCAT, nullptr);
    // GIN[F,800] = W_ih @ e + (b_ih + b_hh)    (f32, no activation)
    k_gemm_wmma<true,  false><<<ggrid(800),  blk, 0, stream>>>(WBih, HCAT + 200, BSUM, 800, 800, 416, 608, 800, 0, nullptr, GIN);

    // --- phase 2: sequential LSTM scan (W_hh resident in 320KB WGP LDS) ---
    constexpr size_t SMEM = 320000 + 1024 + 800 * 4;  // 324224 B <= 320KB LDS
    k_lstm_scan<<<1, 800, SMEM, stream>>>(Whh, GIN, HCAT, F);

    // --- phase 3: frame-parallel fuse + decode + mask ---
    // TB[F,800] = lrelu(Wf1 @ [h;e] + bf1)
    k_gemm_wmma<false, true ><<<ggrid(800),  blk, 0, stream>>>(WBf1, HCAT, bf1, 800, 800, 608, 608, 800, 0, TB, nullptr);
    // T2[F,416] = lrelu(Wf2 @ t + bf2)
    k_gemm_wmma<false, true ><<<ggrid(416),  blk, 0, stream>>>(WBf2, TB, bf2, 400, 416, 800, 800, 416, 0, T2, nullptr);
    // D (reuse E1) = lrelu(W3 @ t2 + b3)
    k_gemm_wmma<false, true ><<<ggrid(1008), blk, 0, stream>>>(WB3, T2, b3, 1000, 1008, 416, 416, 1008, 0, E1, nullptr);
    // out[m,n] = sigmoid(W4 @ d + b4) * X[m,n]
    k_gemm_mask_out<<<ggrid(2064), blk, 0, stream>>>(WB4, E1, b4, X, out, 2049, 2064, 1008, 1008, F);
}
